// DotProductAttention_86766929314140
// MI455X (gfx1250) — matrix operand order repaired and checked by reading, not compiled, not run on hardware
//
#include <hip/hip_runtime.h>

// ---------------------------------------------------------------------------
// Types / helpers for CDNA5 wave32 WMMA (V_WMMA_F32_16X16X32_BF16)
// ---------------------------------------------------------------------------
typedef __attribute__((ext_vector_type(16))) __bf16 v16bf;
typedef __attribute__((ext_vector_type(8)))  float  v8f;

union ABf  { v16bf v; __bf16 e[16]; };
union Chunk { uint4 q; __bf16 b[8]; };

// A-matrix (16x32, 16-bit) per-lane k index for element e, lane-half g.
// ISA 7.12.2: lanes 0-15: VGPR0..3 -> K 0..7, VGPR4..7 -> K 16..23;
//             lanes 16-31: K 8..15 and K 24..31.
// => contiguous runs: e=0..7 -> k=g*8+e ; e=8..15 -> k=16+g*8+(e-8)
__device__ inline int wmma_a_k(int e, int g) {
    return (e < 8) ? (g * 8 + e) : (16 + g * 8 + (e - 8));
}
// B-matrix (32x16, 16-bit): same k order as A: e=0..7 -> k=g*8+e ; e=8..15 -> k=16+g*8+(e-8).
__device__ inline int wmma_b_k(int e, int g) { return (e < 8) ? (g * 8 + e) : (16 + g * 8 + (e - 8)); }

#define WMMA_BF16(A, B, C) \
    __builtin_amdgcn_wmma_f32_16x16x32_bf16(false, (A), false, (B), (short)0, (C), false, false)

#define BB   512
#define HH   16

// ---------------------------------------------------------------------------
// f32 -> bf16 conversion kernel (weights)
// ---------------------------------------------------------------------------
__global__ __launch_bounds__(256) void cvt_f32_bf16(const float* __restrict__ src,
                                                    __bf16* __restrict__ dst, int n) {
    int i = blockIdx.x * 256 + threadIdx.x;
    if (i < n) dst[i] = (__bf16)src[i];
}

// ---------------------------------------------------------------------------
// RPE tables.
//  - Transposed (for QR/KR B-fragments): RT[h][r(256)][d(64)] = rpe[(d*16+h)*256+r]
//  - Direct     (for AR@Rv^T B-fragments): R[h][d(64)][r(256)] = rpe[(d*16+h)*256+r]
//  r >= 225 zeroed (AR GEMM runs K=256 with zero padding).
// ---------------------------------------------------------------------------
__global__ __launch_bounds__(256) void build_rpe_T(const float* __restrict__ rpe,
                                                   __bf16* __restrict__ RT) {
    int i = blockIdx.x * 256 + threadIdx.x;       // [0, 16*256*64)
    if (i >= 16 * 256 * 64) return;
    int d = i & 63;
    int r = (i >> 6) & 255;
    int h = i >> 14;
    RT[i] = (r < 225) ? (__bf16)rpe[(size_t)(d * 16 + h) * 256 + r] : (__bf16)0.f;
}

__global__ __launch_bounds__(256) void build_rpe(const float* __restrict__ rpe,
                                                 __bf16* __restrict__ R) {
    int i = blockIdx.x * 256 + threadIdx.x;       // [0, 16*64*256)
    if (i >= 16 * 64 * 256) return;
    int r = i & 255;
    int d = (i >> 8) & 63;
    int h = i >> 14;
    R[i] = (r < 225) ? (__bf16)rpe[(size_t)(d * 16 + h) * 256 + r] : (__bf16)0.f;
}

// ---------------------------------------------------------------------------
// Generic bf16 WMMA GEMM:  out[m,n] = sum_k A[m,k]*Bw[n,k] + bias[n]
//   A: f32 or bf16 (template) row-major [M,K]; Bw: bf16 row-major [N,K]
//   out: f32 (Of) or bf16 (Ob), row-major [M,N]
// Block = 256 threads (8 waves), tile 128(M) x 64(N), K-step 32.
// Requires: M % 128 == 0, K % 32 == 0. N edge guarded.
// LDS row stride 40 halves (80 B): 16-B aligned rows (b128-mergeable
// fragment loads), conflict-free (20 banks/row, 16-row cycle distinct).
// ---------------------------------------------------------------------------
#define LDA 40
template <bool A_BF16>
__global__ __launch_bounds__(256) void gemm_bf16(const float* __restrict__ Af,
                                                 const __bf16* __restrict__ Ab,
                                                 const __bf16* __restrict__ Bw,
                                                 const float* __restrict__ bias,
                                                 float* __restrict__ Of,
                                                 __bf16* __restrict__ Ob,
                                                 int M, int N, int K) {
    __shared__ __bf16 Asub[128][LDA];
    __shared__ __bf16 Bsub[64][LDA];

    const int tid  = threadIdx.x;
    const int lane = tid & 31;
    const int wave = __builtin_amdgcn_readfirstlane(tid >> 5);   // wave-uniform (SGPR)
    const int g    = lane >> 4;
    const int l16  = lane & 15;

    const int bm = blockIdx.y * 128;
    const int bn = blockIdx.x * 64;
    const int wm = (wave >> 1) * 32;
    const int wn = (wave & 1) * 32;

    v8f vz = {0.f, 0.f, 0.f, 0.f, 0.f, 0.f, 0.f, 0.f};
    v8f acc[2][2];
    acc[0][0] = vz; acc[0][1] = vz; acc[1][0] = vz; acc[1][1] = vz;

    for (int k0 = 0; k0 < K; k0 += 32) {
        // Stage A tile 128x32: 16 elems/thread (contiguous 32 B per thread)
        {
            int r  = tid >> 1;
            int cb = (tid & 1) * 16;
            size_t base = (size_t)(bm + r) * K + k0 + cb;
            if constexpr (A_BF16) {
                #pragma unroll
                for (int j = 0; j < 16; ++j) Asub[r][cb + j] = Ab[base + j];
            } else {
                #pragma unroll
                for (int j = 0; j < 16; ++j) Asub[r][cb + j] = (__bf16)Af[base + j];
            }
        }
        // Stage B tile 64x32: 8 elems/thread, zero-fill n >= N
        {
            int r  = tid >> 2;
            int cb = (tid & 3) * 8;
            int n  = bn + r;
            if (n < N) {
                size_t base = (size_t)n * K + k0 + cb;
                #pragma unroll
                for (int j = 0; j < 8; ++j) Bsub[r][cb + j] = Bw[base + j];
            } else {
                #pragma unroll
                for (int j = 0; j < 8; ++j) Bsub[r][cb + j] = (__bf16)0.f;
            }
        }
        __syncthreads();

        ABf a0, a1, b0, b1;
        #pragma unroll
        for (int e = 0; e < 16; ++e) {
            int ka = wmma_a_k(e, g);
            int kb = wmma_b_k(e, g);
            a0.e[e] = Asub[wm + l16][ka];
            a1.e[e] = Asub[wm + 16 + l16][ka];
            b0.e[e] = Bsub[wn + l16][kb];
            b1.e[e] = Bsub[wn + 16 + l16][kb];
        }
        acc[0][0] = WMMA_BF16(a0.v, b0.v, acc[0][0]);
        acc[0][1] = WMMA_BF16(a0.v, b1.v, acc[0][1]);
        acc[1][0] = WMMA_BF16(a1.v, b0.v, acc[1][0]);
        acc[1][1] = WMMA_BF16(a1.v, b1.v, acc[1][1]);
        __syncthreads();
    }

    // Store C tiles (C layout: row = r + 8*g, col = lane&15)
    #pragma unroll
    for (int i = 0; i < 2; ++i)
        #pragma unroll
        for (int j = 0; j < 2; ++j)
            #pragma unroll
            for (int r = 0; r < 8; ++r) {
                int m = bm + wm + i * 16 + r + 8 * g;
                int n = bn + wn + j * 16 + l16;
                if (n < N) {
                    float v = acc[i][j][r] + (bias ? bias[n] : 0.f);
                    size_t o = (size_t)m * N + n;
                    if (Ob) Ob[o] = (__bf16)v;
                    else    Of[o] = v;
                }
            }
}

// ---------------------------------------------------------------------------
// In-place silu + LayerNorm over `cols` (one block/row; cols%256==0, <=4096)
// ---------------------------------------------------------------------------
__global__ __launch_bounds__(256) void silu_ln_kernel(float* __restrict__ x,
                                                      const float* __restrict__ gw,
                                                      const float* __restrict__ bw,
                                                      int cols) {
    __shared__ float red[256];
    const int tid = threadIdx.x;
    float* p = x + (size_t)blockIdx.x * cols;
    const int per = cols >> 8;
    float v[16];

    float s = 0.f;
    for (int i = 0; i < per; ++i) {
        float t = p[i * 256 + tid];
        t = t / (1.f + __expf(-t));          // silu
        v[i] = t;
        s += t;
    }
    red[tid] = s; __syncthreads();
    for (int o = 128; o > 0; o >>= 1) { if (tid < o) red[tid] += red[tid + o]; __syncthreads(); }
    float mu = red[0] / (float)cols;
    __syncthreads();

    float s2 = 0.f;
    for (int i = 0; i < per; ++i) { float d = v[i] - mu; s2 += d * d; }
    red[tid] = s2; __syncthreads();
    for (int o = 128; o > 0; o >>= 1) { if (tid < o) red[tid] += red[tid + o]; __syncthreads(); }
    float rstd = rsqrtf(red[0] / (float)cols + 1e-5f);
    __syncthreads();

    for (int i = 0; i < per; ++i) {
        int c = i * 256 + tid;
        p[c] = (v[i] - mu) * rstd * gw[c] + bw[c];
    }
}

// ---------------------------------------------------------------------------
// Fused attention per (b,h): scores = (QK^T + Q@Rq gather + K@Rk gather)/8
//                            + smolgen; softmax; H = A@V + AR@Rv^T
// Block = 256 threads (8 waves). LDS = 48KB with AR overlaying Sc/Qs/Ks.
// V is stored transposed in LDS; Rq/Rk tables are pre-transposed in ws so all
// B-fragment reads are 16B-contiguous per lane. Wave tile loops use a
// readfirstlane'd wave id so they compile as scalar (s_branch) loops.
// ---------------------------------------------------------------------------
__global__ __launch_bounds__(256) void attn_kernel(const __bf16* __restrict__ qkv, // [B*S,3072] bf16
                                                   const float*  __restrict__ smol,// [B*H,4096] f32
                                                   const __bf16* __restrict__ RqT, // [H][256][64]
                                                   const __bf16* __restrict__ RkT, // [H][256][64]
                                                   const __bf16* __restrict__ Rv,  // [H][64][256]
                                                   __bf16* __restrict__ hcat) {    // [B*S,1024] bf16
    __shared__ __align__(16) unsigned char smem[49152];
    float*  Sc  = (float*)smem;                       // [64][64] f32, 16KB
    __bf16* Qs  = (__bf16*)(smem + 16384);            // [64][64], 8KB
    __bf16* Ks  = (__bf16*)(smem + 24576);            // [64][64], 8KB
    __bf16* VsT = (__bf16*)(smem + 32768);            // [d][s] transposed, 8KB
    __bf16* As  = (__bf16*)(smem + 40960);            // softmax out, 8KB
    __bf16* AR  = (__bf16*)smem;                      // [64][256] (overlays Sc/Qs/Ks)

    const int b    = blockIdx.x >> 4;
    const int h    = blockIdx.x & 15;
    const int tid  = threadIdx.x;
    const int lane = tid & 31;
    const int wave = __builtin_amdgcn_readfirstlane(tid >> 5);   // wave-uniform (SGPR)
    const int g    = lane >> 4;
    const int l16  = lane & 15;
    const v8f vz   = {0.f, 0.f, 0.f, 0.f, 0.f, 0.f, 0.f, 0.f};

    // --- 1. load Q,K (16B chunks) and V (transposed) ----------------------
    for (int i = tid; i < 64 * 16; i += 256) {        // 1024 chunks of 8 halves
        int s = i >> 4, c = i & 15;
        Chunk ch;
        ch.q = ((const uint4*)(qkv + (size_t)(b * 64 + s) * 3072 + h * 192))[c];
        __bf16* dst = (c < 8) ? (Qs + s * 64 + c * 8) : (Ks + s * 64 + (c - 8) * 8);
        *(uint4*)dst = ch.q;
    }
    for (int i = tid; i < 64 * 8; i += 256) {         // V: 512 chunks, scatter-transpose
        int s = i >> 3, c = i & 7;
        Chunk ch;
        ch.q = ((const uint4*)(qkv + (size_t)(b * 64 + s) * 3072 + h * 192 + 128))[c];
        #pragma unroll
        for (int j = 0; j < 8; ++j) VsT[(c * 8 + j) * 64 + s] = ch.b[j];
    }
    __syncthreads();

    // --- 2. scores = Q @ K^T ----------------------------------------------
    for (int t = wave; t < 16; t += 8) {
        int mt = t >> 2, nt = t & 3;
        v8f acc = vz;
        #pragma unroll
        for (int ks = 0; ks < 2; ++ks) {
            ABf a, bf;
            #pragma unroll
            for (int e = 0; e < 16; ++e) {
                a.e[e]  = Qs[(mt * 16 + l16) * 64 + ks * 32 + wmma_a_k(e, g)];
                bf.e[e] = Ks[(nt * 16 + l16) * 64 + ks * 32 + wmma_b_k(e, g)];
            }
            acc = WMMA_BF16(a.v, bf.v, acc);
        }
        #pragma unroll
        for (int r = 0; r < 8; ++r)
            Sc[(mt * 16 + r + 8 * g) * 64 + nt * 16 + l16] = acc[r];
    }
    __syncthreads();

    // --- 3. QR = Q @ Rq_h, scatter-add into scores -------------------------
    {
        const __bf16* RqTh = RqT + (size_t)h * 256 * 64;
        for (int t = wave; t < 60; t += 8) {          // 4 M-tiles x 15 N-tiles (240 rels)
            int mt = t / 15, nt = t % 15;
            v8f acc = vz;
            #pragma unroll
            for (int ks = 0; ks < 2; ++ks) {
                ABf a, bf;
                #pragma unroll
                for (int e = 0; e < 16; ++e) {
                    a.e[e]  = Qs[(mt * 16 + l16) * 64 + ks * 32 + wmma_a_k(e, g)];
                    bf.e[e] = RqTh[(nt * 16 + l16) * 64 + ks * 32 + wmma_b_k(e, g)];
                }
                acc = WMMA_BF16(a.v, bf.v, acc);
            }
            #pragma unroll
            for (int r = 0; r < 8; ++r) {
                int q   = mt * 16 + r + 8 * g;
                int rel = nt * 16 + l16;
                if (rel < 225) {
                    int ki = (q >> 3) - (rel / 15 - 7);
                    int kj = (q & 7)  - (rel % 15 - 7);
                    if ((unsigned)ki < 8u && (unsigned)kj < 8u)
                        Sc[q * 64 + ki * 8 + kj] += acc[r];   // bijective: race-free
                }
            }
        }
    }
    __syncthreads();

    // --- 4. KR = K @ Rk_h, scatter-add into scores -------------------------
    {
        const __bf16* RkTh = RkT + (size_t)h * 256 * 64;
        for (int t = wave; t < 60; t += 8) {
            int mt = t / 15, nt = t % 15;
            v8f acc = vz;
            #pragma unroll
            for (int ks = 0; ks < 2; ++ks) {
                ABf a, bf;
                #pragma unroll
                for (int e = 0; e < 16; ++e) {
                    a.e[e]  = Ks[(mt * 16 + l16) * 64 + ks * 32 + wmma_a_k(e, g)];
                    bf.e[e] = RkTh[(nt * 16 + l16) * 64 + ks * 32 + wmma_b_k(e, g)];
                }
                acc = WMMA_BF16(a.v, bf.v, acc);
            }
            #pragma unroll
            for (int r = 0; r < 8; ++r) {
                int k   = mt * 16 + r + 8 * g;
                int rel = nt * 16 + l16;
                if (rel < 225) {
                    int qi = (k >> 3) + (rel / 15 - 7);
                    int qj = (k & 7)  + (rel % 15 - 7);
                    if ((unsigned)qi < 8u && (unsigned)qj < 8u)
                        Sc[(qi * 8 + qj) * 64 + k] += acc[r];
                }
            }
        }
    }
    __syncthreads();

    // --- 5. scale + smolgen bias ------------------------------------------
    {
        const float* sm = smol + (size_t)blockIdx.x * 4096;
        for (int i = tid; i < 4096; i += 256) Sc[i] = Sc[i] * 0.125f + sm[i];
    }
    __syncthreads();

    // --- 6. softmax over k (4 lanes per row, wave32 shuffles) -------------
    {
        int q   = tid >> 2;          // 0..63 (groups of 4 lanes stay in-wave)
        int sub = tid & 3;
        float* row = Sc + q * 64;
        float mx = -1e30f;
        float p[16];
        #pragma unroll
        for (int j = 0; j < 16; ++j) mx = fmaxf(mx, row[sub * 16 + j]);
        mx = fmaxf(mx, __shfl_xor(mx, 1, 4));
        mx = fmaxf(mx, __shfl_xor(mx, 2, 4));
        float sum = 0.f;
        #pragma unroll
        for (int j = 0; j < 16; ++j) { p[j] = __expf(row[sub * 16 + j] - mx); sum += p[j]; }
        sum += __shfl_xor(sum, 1, 4);
        sum += __shfl_xor(sum, 2, 4);
        float inv = 1.f / sum;
        #pragma unroll
        for (int j = 0; j < 16; ++j) As[q * 64 + sub * 16 + j] = (__bf16)(p[j] * inv);
    }
    __syncthreads();

    // --- 7. AR[q][rel] = A[q][k]  (Sc/Qs/Ks now dead; AR overlays them) ----
    for (int i = tid; i < 64 * 256; i += 256) AR[i] = (__bf16)0.f;
    __syncthreads();
    for (int i = tid; i < 4096; i += 256) {
        int q = i >> 6, k = i & 63;
        int rel = 15 * ((q >> 3) - (k >> 3) + 7) + ((q & 7) - (k & 7) + 7);
        AR[q * 256 + rel] = As[i];   // injective per q
    }
    __syncthreads();

    // --- 8. Hout = A @ V + AR @ Rv^T --------------------------------------
    {
        const __bf16* Rvh = Rv + (size_t)h * 64 * 256;
        for (int t = wave; t < 16; t += 8) {
            int mt = t >> 2, nt = t & 3;
            v8f acc = vz;
            #pragma unroll
            for (int ks = 0; ks < 2; ++ks) {           // A @ V  (K = 64), V transposed
                ABf a, bf;
                #pragma unroll
                for (int e = 0; e < 16; ++e) {
                    a.e[e]  = As[(mt * 16 + l16) * 64 + ks * 32 + wmma_a_k(e, g)];
                    bf.e[e] = VsT[(nt * 16 + l16) * 64 + ks * 32 + wmma_b_k(e, g)];
                }
                acc = WMMA_BF16(a.v, bf.v, acc);
            }
            #pragma unroll
            for (int ks = 0; ks < 8; ++ks) {           // AR @ Rv^T (K = 256, zero-padded)
                ABf a, bf;
                #pragma unroll
                for (int e = 0; e < 16; ++e) {
                    a.e[e]  = AR[(mt * 16 + l16) * 256 + ks * 32 + wmma_a_k(e, g)];
                    bf.e[e] = Rvh[(nt * 16 + l16) * 256 + ks * 32 + wmma_b_k(e, g)];
                }
                acc = WMMA_BF16(a.v, bf.v, acc);
            }
            #pragma unroll
            for (int r = 0; r < 8; ++r) {
                int q = mt * 16 + r + 8 * g;
                int d = nt * 16 + l16;
                hcat[(size_t)(b * 64 + q) * 1024 + h * 64 + d] = (__bf16)acc[r];
            }
        }
    }
}

// ---------------------------------------------------------------------------
// Host-side orchestration
// ---------------------------------------------------------------------------
extern "C" void kernel_launch(void* const* d_in, const int* in_sizes, int n_in,
                              void* d_out, int out_size, void* d_ws, size_t ws_size,
                              hipStream_t stream) {
    (void)in_sizes; (void)n_in; (void)out_size; (void)ws_size;

    const float* x      = (const float*)d_in[0];
    const float* query  = (const float*)d_in[1];
    const float* W_qkv  = (const float*)d_in[5];
    const float* W_h_w  = (const float*)d_in[6];
    const float* W_h_b  = (const float*)d_in[7];
    const float* rpe_q  = (const float*)d_in[8];
    const float* rpe_k  = (const float*)d_in[9];
    const float* rpe_v  = (const float*)d_in[10];
    const float* sm1_w  = (const float*)d_in[11];
    const float* sm1_b  = (const float*)d_in[12];
    const float* sm2_w  = (const float*)d_in[13];
    const float* sm2_b  = (const float*)d_in[14];
    const float* ln1_g  = (const float*)d_in[15];
    const float* ln1_b  = (const float*)d_in[16];
    const float* sm3_w  = (const float*)d_in[17];
    const float* sm3_b  = (const float*)d_in[18];
    const float* ln2_g  = (const float*)d_in[19];
    const float* ln2_b  = (const float*)d_in[20];
    const float* prep_w = (const float*)d_in[21];
    const float* prep_b = (const float*)d_in[22];
    float* out = (float*)d_out;

    // Workspace carve (256B aligned)
    char* ws = (char*)d_ws;
    size_t off = 0;
    auto carve = [&](size_t bytes) { size_t r = off; off += (bytes + 255) & ~(size_t)255; return r; };

    __bf16* qkv_bf  = (__bf16*)(ws + carve((size_t)32768 * 3072 * 2));
    float*  sg1     = (float*)(ws + carve((size_t)32768 * 32 * 4));
    float*  sg2     = (float*)(ws + carve((size_t)512 * 256 * 4));
    float*  sg3     = (float*)(ws + carve((size_t)512 * 4096 * 4));
    float*  smol    = (float*)(ws + carve((size_t)8192 * 4096 * 4));
    __bf16* hcat    = (__bf16*)(ws + carve((size_t)32768 * 1024 * 2));
    __bf16* wqkv_bf = (__bf16*)(ws + carve((size_t)3072 * 1024 * 2));
    __bf16* whw_bf  = (__bf16*)(ws + carve((size_t)1024 * 1024 * 2));
    __bf16* sm1w_bf = (__bf16*)(ws + carve((size_t)32 * 1024 * 2));
    __bf16* sm2w_bf = (__bf16*)(ws + carve((size_t)256 * 2048 * 2));
    __bf16* sm3w_bf = (__bf16*)(ws + carve((size_t)4096 * 256 * 2));
    __bf16* prpw_bf = (__bf16*)(ws + carve((size_t)4096 * 256 * 2));
    __bf16* RqT     = (__bf16*)(ws + carve((size_t)16 * 256 * 64 * 2));
    __bf16* RkT     = (__bf16*)(ws + carve((size_t)16 * 256 * 64 * 2));
    __bf16* Rv      = (__bf16*)(ws + carve((size_t)16 * 64 * 256 * 2));

    auto cvt = [&](const float* s, __bf16* d, int n) {
        cvt_f32_bf16<<<(n + 255) / 256, 256, 0, stream>>>(s, d, n);
    };
    cvt(W_qkv,  wqkv_bf, 3072 * 1024);
    cvt(W_h_w,  whw_bf,  1024 * 1024);
    cvt(sm1_w,  sm1w_bf, 32 * 1024);
    cvt(sm2_w,  sm2w_bf, 256 * 2048);
    cvt(sm3_w,  sm3w_bf, 4096 * 256);
    cvt(prep_w, prpw_bf, 4096 * 256);

    build_rpe_T<<<(16 * 256 * 64) / 256, 256, 0, stream>>>(rpe_q, RqT);
    build_rpe_T<<<(16 * 256 * 64) / 256, 256, 0, stream>>>(rpe_k, RkT);
    build_rpe  <<<(16 * 64 * 256) / 256, 256, 0, stream>>>(rpe_v, Rv);

    // QKV projection: [32768,1024] x [3072,1024]^T -> bf16 [32768,3072]
    gemm_bf16<false><<<dim3(3072 / 64, 32768 / 128), 256, 0, stream>>>(
        query, nullptr, wqkv_bf, nullptr, nullptr, qkv_bf, 32768, 3072, 1024);

    // smolgen stage 1: [32768,1024] x [32,1024]^T + b -> f32 [32768,32]
    gemm_bf16<false><<<dim3(1, 32768 / 128), 256, 0, stream>>>(
        x, nullptr, sm1w_bf, sm1_b, sg1, nullptr, 32768, 32, 1024);

    // stage 2: [512,2048] x [256,2048]^T + b -> [512,256]
    gemm_bf16<false><<<dim3(256 / 64, 512 / 128), 256, 0, stream>>>(
        sg1, nullptr, sm2w_bf, sm2_b, sg2, nullptr, 512, 256, 2048);
    silu_ln_kernel<<<512, 256, 0, stream>>>(sg2, ln1_g, ln1_b, 256);

    // stage 3: [512,256] x [4096,256]^T + b -> [512,4096]
    gemm_bf16<false><<<dim3(4096 / 64, 512 / 128), 256, 0, stream>>>(
        sg2, nullptr, sm3w_bf, sm3_b, sg3, nullptr, 512, 4096, 256);
    silu_ln_kernel<<<512, 256, 0, stream>>>(sg3, ln2_g, ln2_b, 4096);

    // prep: [8192,256] x [4096,256]^T + b -> smolgen bias [8192,4096]
    gemm_bf16<false><<<dim3(4096 / 64, 8192 / 128), 256, 0, stream>>>(
        sg3, nullptr, prpw_bf, prep_b, smol, nullptr, 8192, 4096, 256);

    // fused attention per (b,h)
    attn_kernel<<<BB * HH, 256, 0, stream>>>(qkv_bf, smol, RqT, RkT, Rv, hcat);

    // output projection: [32768,1024] x [1024,1024]^T + b -> f32 out
    gemm_bf16<true><<<dim3(1024 / 64, 32768 / 128), 256, 0, stream>>>(
        nullptr, hcat, whw_bf, W_h_b, out, nullptr, 32768, 1024, 1024);
}
